// GATV_4260607557873
// MI455X (gfx1250) — matrix operand order repaired and checked by reading, not compiled, not run on hardware
//
#include <hip/hip_runtime.h>
#include <hip/hip_bf16.h>
#include <stdint.h>

typedef __attribute__((ext_vector_type(16))) _Float16 v16h;
typedef __attribute__((ext_vector_type(8)))  float    v8f;
typedef __attribute__((ext_vector_type(4)))  unsigned int u32x4;
typedef __attribute__((ext_vector_type(8)))  int      i32x8;
typedef __attribute__((ext_vector_type(4)))  int      i32x4;

#define AS_LDS    __attribute__((address_space(3)))

#define N_NODES 8192
#define FIN     64
#define HID     64
#define HEADS   2
#define COUT    16
#define CSPLIT  4
#define SEG     (N_NODES / CSPLIT)   /* 2048 columns per block */
#define ALPHA   0.2f
#define LOG2E   1.44269504088896f
#define LN2     0.693147180559945f

#if __has_builtin(__builtin_amdgcn_global_load_async_to_lds_b128) && \
    __has_builtin(__builtin_amdgcn_s_wait_asynccnt)
#define HAVE_ASYNC 1
#else
#define HAVE_ASYNC 0
#endif

#if __has_builtin(__builtin_amdgcn_tensor_load_to_lds) && \
    __has_builtin(__builtin_amdgcn_s_wait_tensorcnt)
#define HAVE_TDM 1
#else
#define HAVE_TDM 0
#endif

#if HAVE_ASYNC
__device__ __forceinline__ void async_b128(const void* g, void* l)
{
    // Toolchain declares params as generic pointers to v4i.
    __builtin_amdgcn_global_load_async_to_lds_b128(
        (i32x4*)(uintptr_t)g, (i32x4*)l, 0, 0);
}
#endif

// ---------------------------------------------------------------------------
// Stage 1: Wh = x @ W per head (f16 transposed copy for WMMA-B), s = Wh@a_lo,
// t = Wh@a_hi.  Tiny GEMM (134 MFLOP) -> plain VALU with W in LDS.
// ---------------------------------------------------------------------------
__global__ void k_prep1(const float* __restrict__ x,
                        const float* __restrict__ W_all,   // [H][FIN][HID]
                        const float* __restrict__ a_all,   // [H][2*HID]
                        _Float16* __restrict__ WhT16,      // [H][HID][N]
                        float* __restrict__ s1,            // [H][N]
                        float* __restrict__ t1)            // [H][N]
{
    __shared__ float Wlds[FIN * HID];
    __shared__ float alds[2 * HID];
    const int head = blockIdx.y;
    const int tid  = threadIdx.x;
    for (int i = tid; i < FIN * HID; i += 256) Wlds[i] = W_all[head * FIN * HID + i];
    if (tid < 2 * HID) alds[tid] = a_all[head * 2 * HID + tid];
    __syncthreads();

    const int row = blockIdx.x * 256 + tid;
    float xr[FIN];
#pragma unroll
    for (int k = 0; k < FIN; k += 4) {
        float4 v = *(const float4*)(x + (size_t)row * FIN + k);
        xr[k] = v.x; xr[k + 1] = v.y; xr[k + 2] = v.z; xr[k + 3] = v.w;
    }
    float s = 0.f, t = 0.f;
    for (int f = 0; f < HID; ++f) {
        float acc = 0.f;
#pragma unroll
        for (int k = 0; k < FIN; ++k) acc = fmaf(xr[k], Wlds[k * HID + f], acc);
        WhT16[(size_t)(head * HID + f) * N_NODES + row] = (_Float16)acc;
        s = fmaf(acc, alds[f], s);
        t = fmaf(acc, alds[HID + f], t);
    }
    s1[head * N_NODES + row] = s;
    t1[head * N_NODES + row] = t;
}

// ---------------------------------------------------------------------------
// max over t (per head) -> safe softmax shift m_i = leakyrelu(s_i + max_t)
// ---------------------------------------------------------------------------
__global__ void k_maxt(const float* __restrict__ t, float* __restrict__ mx, int n)
{
    __shared__ float red[256];
    const int tid = threadIdx.x;
    const float* tp = t + (size_t)blockIdx.x * n;
    float m = -3.4e38f;
    for (int i = tid; i < n; i += 256) m = fmaxf(m, tp[i]);
    red[tid] = m;
    __syncthreads();
    for (int sft = 128; sft > 0; sft >>= 1) {
        if (tid < sft) red[tid] = fmaxf(red[tid], red[tid + sft]);
        __syncthreads();
    }
    if (tid == 0) mx[blockIdx.x] = red[0];
}

// ---------------------------------------------------------------------------
// Fused masked-softmax-attention:  out_partial = P @ Wh,  l_partial = P @ 1
// One wave owns a 16-row block; 8 waves (128 rows) share LDS B tiles.
// B tiles double-buffered via GLOBAL_LOAD_ASYNC_TO_LDS (ASYNCcnt), t segment
// staged once via the Tensor Data Mover (TENSORcnt) when available.
// ---------------------------------------------------------------------------
__device__ __forceinline__ _Float16 patt(int ad, float tj, float si, float mlog)
{
    float e  = si + tj;
    float le = fmaxf(e, 0.f) + ALPHA * fminf(e, 0.f);
    float p  = __builtin_amdgcn_exp2f(fmaf(le, LOG2E, -mlog));   // <= 1.0
    return (_Float16)(ad != 0 ? p : 0.f);
}

template <int FB>
__device__ __forceinline__ void stageB_sync(const _Float16* __restrict__ WhTh,
                                            int colBase, int kt,
                                            _Float16* __restrict__ dst, int tid)
{
    if (tid < FB * 16 * 4) {
        int f = tid >> 2, c = tid & 3;
        uint4 v = *(const uint4*)(WhTh + (size_t)f * N_NODES + colBase + kt + c * 8);
        *(uint4*)&dst[f * 32 + c * 8] = v;
    }
}

#if HAVE_ASYNC
template <int FB>
__device__ __forceinline__ void stageB_async(const _Float16* __restrict__ WhTh,
                                             int colBase, int kt,
                                             _Float16* __restrict__ dst, int tid)
{
    if (tid < FB * 16 * 4) {
        int f = tid >> 2, c = tid & 3;
        async_b128(WhTh + (size_t)f * N_NODES + colBase + kt + c * 8,
                   dst + f * 32 + c * 8);
    }
}
#endif

template <int FB>
__global__ void __launch_bounds__(256)
k_attn(const int* __restrict__ adj,          // [N][N]
       const _Float16* __restrict__ WhT,     // [H][FOUT][N]  (transposed, f16)
       const float* __restrict__ sA,         // [H][N]
       const float* __restrict__ tA,         // [H][N]
       const float* __restrict__ mtA,        // [H]
       float* __restrict__ accP,             // [H][CSPLIT][N][FOUT]
       float* __restrict__ lP)               // [H][CSPLIT][N]
{
    constexpr int FOUT = FB * 16;
    __shared__ float tl[SEG];
    __shared__ __align__(32) _Float16 bl[2][FOUT * 32];   // B tile, [F][K] f16

    const int tid  = threadIdx.x;
    const int wave = tid >> 5;
    const int lane = tid & 31;
    const int hi   = (lane >> 4) & 1;
    const int n    = lane & 15;
    const int h    = blockIdx.z;
    const int cs   = blockIdx.y;
    const int r0   = blockIdx.x * 128 + wave * 16;
    const int colBase = cs * SEG;

    const _Float16* WhTh = WhT + (size_t)h * FOUT * N_NODES;

    // ---- stage t segment into LDS ------------------------------------------
#if HAVE_TDM
    if (wave == 0) {
        // 2D D#: tile 2048x1 of 4-byte elements out of an 8192-wide tensor.
        const float* gsrc = tA + (size_t)h * N_NODES + colBase;
        uint64_t ga = (uint64_t)(uintptr_t)gsrc;
        uint32_t la = (uint32_t)(uintptr_t)(AS_LDS void*)tl;
        u32x4 g0;
        g0[0] = 1u;                                             // count=1
        g0[1] = la;                                             // lds_addr
        g0[2] = (uint32_t)ga;                                   // global_addr lo
        g0[3] = ((uint32_t)(ga >> 32) & 0x01FFFFFFu) | (2u << 30); // hi + type=2
        i32x8 g1;
        g1[0] = (int)(2u << 16);                 // data_size=2 (4B)
        g1[1] = (int)((N_NODES & 0xFFFFu) << 16);// tensor_dim0 lo16 in [31:16]
        g1[2] = (int)((N_NODES >> 16) | (1u << 16)); // dim0 hi | tensor_dim1=1
        g1[3] = (int)((unsigned)SEG << 16);      // tile_dim0 = 2048
        g1[4] = 1;                               // tile_dim1 = 1
        g1[5] = N_NODES;                         // tensor_dim0_stride lo32
        g1[6] = 0;
        g1[7] = 0;
        i32x4 g2 = (i32x4)0, g3 = (i32x4)0;
#if __clang_major__ >= 23
        i32x8 g4 = (i32x8)0;
        __builtin_amdgcn_tensor_load_to_lds(g0, g1, g2, g3, g4, 0);
#else
        __builtin_amdgcn_tensor_load_to_lds(g0, g1, g2, g3, 0);
#endif
        __builtin_amdgcn_s_wait_tensorcnt(0);
    }
#else
    for (int i = tid; i < SEG / 4; i += 256)
        ((float4*)tl)[i] = ((const float4*)(tA + (size_t)h * N_NODES + colBase))[i];
#endif

    const float si = sA[(size_t)h * N_NODES + r0 + n];
    const float mt = mtA[h];
    const float e0 = si + mt;
    const float m  = fmaxf(e0, 0.f) + ALPHA * fminf(e0, 0.f);
    const float mlog = m * LOG2E;

    const int* arowBase = adj + (size_t)(r0 + n) * N_NODES + colBase + hi * 16;

    v8f acc[FB];
#pragma unroll
    for (int fb = 0; fb < FB; ++fb) acc[fb] = (v8f)(0.f);
    v8f lsum = (v8f)(0.f);
    v16h ones;
#pragma unroll
    for (int i = 0; i < 16; ++i) ones[i] = (_Float16)1.0f;

#if HAVE_ASYNC
    stageB_async<FB>(WhTh, colBase, 0, bl[0], tid);   // prime the pipeline
#endif
    __syncthreads();   // t staged

#pragma unroll 2
    for (int kt = 0; kt < SEG; kt += 32) {
        const int cur = (kt >> 5) & 1;
#if HAVE_ASYNC
        __builtin_amdgcn_s_wait_asynccnt(0);
        __syncthreads();   // bl[cur] visible to all; prior readers of bl[cur^1] done
        if (kt + 32 < SEG)
            stageB_async<FB>(WhTh, colBase, kt + 32, bl[cur ^ 1], tid);
#else
        stageB_sync<FB>(WhTh, colBase, kt, bl[0], tid);
        __syncthreads();
#endif
        const _Float16* blc = HAVE_ASYNC ? bl[cur] : bl[0];

        const int* ar = arowBase + kt;
        int4 a0 = ((const int4*)ar)[0];
        int4 a1 = ((const int4*)ar)[1];
        int4 a2 = ((const int4*)(ar + 8))[0];
        int4 a3 = ((const int4*)(ar + 8))[1];
        __builtin_prefetch(ar + 32, 0, 1);     // next adj tile -> global_prefetch

        const float* tb = tl + kt + hi * 16;
        float4 t0 = ((const float4*)tb)[0];
        float4 t1v = ((const float4*)tb)[1];
        float4 t2v = ((const float4*)(tb + 8))[0];
        float4 t3v = ((const float4*)(tb + 8))[1];

        v16h A;
        A[0]  = patt(a0.x, t0.x,  si, mlog);
        A[1]  = patt(a0.y, t0.y,  si, mlog);
        A[2]  = patt(a0.z, t0.z,  si, mlog);
        A[3]  = patt(a0.w, t0.w,  si, mlog);
        A[4]  = patt(a1.x, t1v.x, si, mlog);
        A[5]  = patt(a1.y, t1v.y, si, mlog);
        A[6]  = patt(a1.z, t1v.z, si, mlog);
        A[7]  = patt(a1.w, t1v.w, si, mlog);
        A[8]  = patt(a2.x, t2v.x, si, mlog);
        A[9]  = patt(a2.y, t2v.y, si, mlog);
        A[10] = patt(a2.z, t2v.z, si, mlog);
        A[11] = patt(a2.w, t2v.w, si, mlog);
        A[12] = patt(a3.x, t3v.x, si, mlog);
        A[13] = patt(a3.y, t3v.y, si, mlog);
        A[14] = patt(a3.z, t3v.z, si, mlog);
        A[15] = patt(a3.w, t3v.w, si, mlog);

#pragma unroll
        for (int fb = 0; fb < FB; ++fb) {
            v16h B = *(const v16h*)&blc[(fb * 16 + n) * 32 + hi * 16];
            acc[fb] = __builtin_amdgcn_wmma_f32_16x16x32_f16(
                false, A, false, B, (short)0, acc[fb], false, false);
        }
        lsum = __builtin_amdgcn_wmma_f32_16x16x32_f16(
            false, A, false, ones, (short)0, lsum, false, false);
#if !HAVE_ASYNC
        __syncthreads();
#endif
    }

    // C layout: VGPR v -> row (v + 8*hi), column n.  lsum holds l replicated.
    float* accOut = accP + (((size_t)h * CSPLIT + cs) * N_NODES + r0) * FOUT;
#pragma unroll
    for (int fb = 0; fb < FB; ++fb)
#pragma unroll
        for (int v = 0; v < 8; ++v)
            accOut[(size_t)(v + 8 * hi) * FOUT + fb * 16 + n] = acc[fb][v];
    if (n == 0) {
        float* lOut = lP + ((size_t)h * CSPLIT + cs) * N_NODES + r0;
#pragma unroll
        for (int v = 0; v < 8; ++v) lOut[v + 8 * hi] = lsum[v];
    }
}

// ---------------------------------------------------------------------------
// Combine layer-1 partials, normalize, concat heads, Whc = h @ W_out,
// s2/t2, and transposed f16 copy of Whc for the output-layer WMMA-B.
// ---------------------------------------------------------------------------
__global__ void k_mid(const float* __restrict__ accP1, const float* __restrict__ lP1,
                      const float* __restrict__ Wout, const float* __restrict__ aout,
                      _Float16* __restrict__ WhT2, float* __restrict__ s2,
                      float* __restrict__ t2)
{
    __shared__ float Wlds[HEADS * HID * COUT];
    __shared__ float alds[2 * COUT];
    const int tid = threadIdx.x;
    for (int i = tid; i < HEADS * HID * COUT; i += 256) Wlds[i] = Wout[i];
    if (tid < 2 * COUT) alds[tid] = aout[tid];
    __syncthreads();

    const int row = blockIdx.x * 256 + tid;
    float whc[COUT];
#pragma unroll
    for (int c = 0; c < COUT; ++c) whc[c] = 0.f;

    for (int h = 0; h < HEADS; ++h) {
        float l = 0.f;
        for (int cs = 0; cs < CSPLIT; ++cs)
            l += lP1[((size_t)h * CSPLIT + cs) * N_NODES + row];
        float invl = 1.0f / l;
        for (int f = 0; f < HID; ++f) {
            float num = 0.f;
            for (int cs = 0; cs < CSPLIT; ++cs)
                num += accP1[(((size_t)h * CSPLIT + cs) * N_NODES + row) * HID + f];
            float hv = num * invl;
            const int k = h * HID + f;
#pragma unroll
            for (int c = 0; c < COUT; ++c) whc[c] = fmaf(hv, Wlds[k * COUT + c], whc[c]);
        }
    }
    float s = 0.f, t = 0.f;
#pragma unroll
    for (int c = 0; c < COUT; ++c) {
        WhT2[(size_t)c * N_NODES + row] = (_Float16)whc[c];
        s = fmaf(whc[c], alds[c], s);
        t = fmaf(whc[c], alds[COUT + c], t);
    }
    s2[row] = s;
    t2[row] = t;
}

// ---------------------------------------------------------------------------
// Combine output-layer partials, normalize, ELU, log_softmax over 16 classes.
// ---------------------------------------------------------------------------
__global__ void k_final(const float* __restrict__ accP2, const float* __restrict__ lP2,
                        float* __restrict__ out)
{
    const int row = blockIdx.x * 256 + threadIdx.x;
    float l = 0.f;
    for (int cs = 0; cs < CSPLIT; ++cs) l += lP2[(size_t)cs * N_NODES + row];
    float invl = 1.0f / l;

    float v[COUT];
    float mx = -3.4e38f;
#pragma unroll
    for (int c = 0; c < COUT; ++c) {
        float num = 0.f;
        for (int cs = 0; cs < CSPLIT; ++cs)
            num += accP2[((size_t)cs * N_NODES + row) * COUT + c];
        float hv = num * invl;
        hv = hv > 0.f ? hv : (__builtin_amdgcn_exp2f(hv * LOG2E) - 1.f);  // ELU
        v[c] = hv;
        mx = fmaxf(mx, hv);
    }
    float sum = 0.f;
#pragma unroll
    for (int c = 0; c < COUT; ++c)
        sum += __builtin_amdgcn_exp2f((v[c] - mx) * LOG2E);
    float lse = __builtin_amdgcn_logf(sum) * LN2 + mx;   // logf is log2 -> *ln2
#pragma unroll
    for (int c = 0; c < COUT; ++c) out[(size_t)row * COUT + c] = v[c] - lse;
}

// ---------------------------------------------------------------------------
extern "C" void kernel_launch(void* const* d_in, const int* in_sizes, int n_in,
                              void* d_out, int out_size, void* d_ws, size_t ws_size,
                              hipStream_t stream)
{
    const float* x      = (const float*)d_in[0];
    const int*   adj    = (const int*)d_in[1];
    const float* Wheads = (const float*)d_in[2];
    const float* aheads = (const float*)d_in[3];
    const float* Wout   = (const float*)d_in[4];
    const float* aout   = (const float*)d_in[5];
    float* out = (float*)d_out;

    char* ws = (char*)d_ws;
    size_t off = 0;
    auto alloc = [&](size_t bytes) -> void* {
        void* p = ws + off;
        off += bytes;
        off = (off + 255) & ~(size_t)255;
        return p;
    };

    _Float16* WhT16 = (_Float16*)alloc((size_t)HEADS * HID * N_NODES * 2);
    float* s1    = (float*)alloc((size_t)HEADS * N_NODES * 4);
    float* t1    = (float*)alloc((size_t)HEADS * N_NODES * 4);
    float* mt1   = (float*)alloc(64);
    float* accP1 = (float*)alloc((size_t)HEADS * CSPLIT * N_NODES * HID * 4);
    float* lP1   = (float*)alloc((size_t)HEADS * CSPLIT * N_NODES * 4);
    _Float16* WhT2 = (_Float16*)alloc((size_t)COUT * N_NODES * 2);
    float* s2    = (float*)alloc((size_t)N_NODES * 4);
    float* t2    = (float*)alloc((size_t)N_NODES * 4);
    float* mt2   = (float*)alloc(64);
    float* accP2 = (float*)alloc((size_t)CSPLIT * N_NODES * COUT * 4);
    float* lP2   = (float*)alloc((size_t)CSPLIT * N_NODES * 4);

    k_prep1<<<dim3(N_NODES / 256, HEADS), 256, 0, stream>>>(x, Wheads, aheads,
                                                            WhT16, s1, t1);
    k_maxt<<<HEADS, 256, 0, stream>>>(t1, mt1, N_NODES);
    k_attn<4><<<dim3(N_NODES / 128, CSPLIT, HEADS), 256, 0, stream>>>(
        adj, WhT16, s1, t1, mt1, accP1, lP1);
    k_mid<<<N_NODES / 256, 256, 0, stream>>>(accP1, lP1, Wout, aout, WhT2, s2, t2);
    k_maxt<<<1, 256, 0, stream>>>(t2, mt2, N_NODES);
    k_attn<1><<<dim3(N_NODES / 128, CSPLIT, 1), 256, 0, stream>>>(
        adj, WhT2, s2, t2, mt2, accP2, lP2);
    k_final<<<N_NODES / 256, 256, 0, stream>>>(accP2, lP2, out);

    (void)in_sizes; (void)n_in; (void)out_size; (void)ws_size;
}
